// GroupNet_49898930045034
// MI455X (gfx1250) — matrix operand order repaired and checked by reading, not compiled, not run on hardware
//
#include <hip/hip_runtime.h>
#include <stdint.h>

typedef __bf16 bf16;
typedef __attribute__((ext_vector_type(16))) __bf16  v16bf;
typedef __attribute__((ext_vector_type(8)))  __bf16  v8bf;
typedef __attribute__((ext_vector_type(8)))  float   v8f;
typedef __attribute__((ext_vector_type(8)))  uint32_t v8u;
typedef __attribute__((ext_vector_type(4)))  int      v4i;

#define BATCH 4096
#define NNODE 128
#define FDIM  28
#define HID   512
#define XROW  3620      // 36 + 128*28
#define NROWS (BATCH*NNODE)

// ---- workspace layout (bytes) ----
#define WS_ASSIGN 0u          // 524288  u8   : per-(b,n) group id 0..7
#define WS_OGE    524288u     // 4096*112 f32 : ogroup_em, [b][g*28+f]
#define WS_W1REP  2359296u    // 32*32*16 bf16: W1 B-fragments
#define WS_W2REP  2392064u    // 16*32*16 bf16: W2 (padded to 16 cols)
#define WS_K2REP  2408448u    // 5*32*32*16 bf16: [self|ogroup] Wa rows
#define WS_WAREP  2572288u    // 4*128*32*32*16 bf16: igroup Wa, K padded 28->32

// ---- CDNA5 async global->LDS copy (ASYNCcnt path), with safe fallback ----
#if defined(__has_builtin) && __has_builtin(__builtin_amdgcn_global_load_async_to_lds_b128)
#define HAVE_ASYNC 1
#define ASYNC_CP16(ldsp, gmem)                                             \
  __builtin_amdgcn_global_load_async_to_lds_b128(                          \
      (__attribute__((address_space(1))) v4i*)(gmem),                      \
      (__attribute__((address_space(3))) v4i*)(ldsp), 0, 0)
#if __has_builtin(__builtin_amdgcn_s_wait_asynccnt)
#define ASYNC_WAIT() __builtin_amdgcn_s_wait_asynccnt(0)
#else
#define ASYNC_WAIT() asm volatile("s_wait_asynccnt 0x0" ::: "memory")
#endif
#else
#define HAVE_ASYNC 0
#define ASYNC_CP16(ldsp, gmem)                                             \
  do { *(uint4*)(ldsp) = *(const uint4*)(gmem); } while (0)
#define ASYNC_WAIT()
#endif

__device__ inline v8f wmma_bf16(v16bf a, v16bf b, v8f c) {
  return __builtin_amdgcn_wmma_f32_16x16x32_bf16(false, a, false, b, (short)0, c,
                                                 false, false);
}

// A-fragment (16x32 bf16) loader from an o-row (28 valid floats, K padded to 32).
// ISA layout: lane<16 holds K {0..7,16..23}; lane>=16 holds K {8..15,24..31}.
__device__ inline v16bf load_o_frag(const float* base, int half) {
  const float* bp = base + (half ? 8 : 0);
  float4 q0 = *(const float4*)(bp + 0);
  float4 q1 = *(const float4*)(bp + 4);
  float4 q2 = *(const float4*)(bp + 16);
  float4 q3 = half ? make_float4(0.f, 0.f, 0.f, 0.f) : *(const float4*)(bp + 20);
  v16bf a;
  a[0]=(bf16)q0.x; a[1]=(bf16)q0.y; a[2]=(bf16)q0.z; a[3]=(bf16)q0.w;
  a[4]=(bf16)q1.x; a[5]=(bf16)q1.y; a[6]=(bf16)q1.z; a[7]=(bf16)q1.w;
  a[8]=(bf16)q2.x; a[9]=(bf16)q2.y; a[10]=(bf16)q2.z; a[11]=(bf16)q2.w;
  a[12]=(bf16)q3.x; a[13]=(bf16)q3.y; a[14]=(bf16)q3.z; a[15]=(bf16)q3.w;
  return a;
}

// ---- JAX threefry2x32 gumbel, key = PRNGKey(42) = [0,42], shape 4096*128*8 ----
__device__ inline uint32_t rotl32(uint32_t x, int r) { return (x << r) | (x >> (32 - r)); }
__device__ inline float gumbel_at(uint32_t idx) {
  const uint32_t HALF = 2097152u;  // (4096*128*8)/2
  uint32_t j = (idx < HALF) ? idx : idx - HALF;
  uint32_t k0 = 0u, k1 = 42u, k2 = k0 ^ k1 ^ 0x1BD11BDAu;
  uint32_t ks[3] = {k0, k1, k2};
  uint32_t x0 = j + k0, x1 = (j + HALF) + k1;
  const int rot[2][4] = {{13, 15, 26, 6}, {17, 29, 16, 24}};
#pragma unroll
  for (int i = 0; i < 5; i++) {
#pragma unroll
    for (int q = 0; q < 4; q++) { x0 += x1; x1 = rotl32(x1, rot[i & 1][q]); x1 ^= x0; }
    x0 += ks[(i + 1) % 3];
    x1 += ks[(i + 2) % 3] + (uint32_t)(i + 1);
  }
  uint32_t bits = (idx < HALF) ? x0 : x1;
  float u = __uint_as_float((bits >> 9) | 0x3f800000u) - 1.0f;
  const float tiny = 1.17549435e-38f;
  u = fmaxf(tiny, u * (1.0f - tiny) + tiny);
  return -__logf(-__logf(u));
}

// ================= repack kernels (weights -> bf16 B-fragment layout) ==========
__global__ __launch_bounds__(256) void repack_w1(const float* W1, bf16* rep) {
  int idx = blockIdx.x * 256 + threadIdx.x;
  if (idx >= 32 * 32 * 16) return;
  int e = idx & 15, lane = (idx >> 4) & 31, j = idx >> 9;
  int k = ((e < 8) ? e : 8 + e) + ((lane < 16) ? 0 : 8);
  int col = j * 16 + (lane & 15);
  rep[idx] = (bf16)((k < FDIM) ? W1[k * HID + col] : 0.0f);
}

__global__ __launch_bounds__(256) void repack_w2(const float* W2, bf16* rep) {
  int idx = blockIdx.x * 256 + threadIdx.x;
  if (idx >= 16 * 32 * 16) return;
  int e = idx & 15, lane = (idx >> 4) & 31, kt = idx >> 9;
  int k = kt * 32 + ((e < 8) ? e : 8 + e) + ((lane < 16) ? 0 : 8);
  int col = lane & 15;
  rep[idx] = (bf16)((col < 8) ? W2[k * 8 + col] : 0.0f);
}

__global__ __launch_bounds__(256) void repack_k2(const float* Wa, bf16* rep) {
  int idx = blockIdx.x * 256 + threadIdx.x;
  if (idx >= 5 * 32 * 32 * 16) return;
  int e = idx & 15, lane = (idx >> 4) & 31, j = (idx >> 9) & 31, kt = idx >> 14;
  int k = kt * 32 + ((e < 8) ? e : 8 + e) + ((lane < 16) ? 0 : 8);
  int col = j * 16 + (lane & 15);
  float v = 0.0f;
  if (k < 36) v = Wa[k * HID + col];
  else if (k < 148) v = Wa[(14372 + (k - 36)) * HID + col];
  rep[idx] = (bf16)v;
}

__global__ __launch_bounds__(256) void repack_wa(const float* Wa, bf16* rep) {
  int idx = blockIdx.x * 256 + threadIdx.x;  // total 4*128*32*32*16 = 8388608
  int e = idx & 15, lane = (idx >> 4) & 31, j = (idx >> 9) & 31;
  int n = (idx >> 14) & 127, g = (idx >> 21) & 3;
  int k = ((e < 8) ? e : 8 + e) + ((lane < 16) ? 0 : 8);
  int col = j * 16 + (lane & 15);
  float v = (k < FDIM) ? Wa[(36 + g * (NNODE * FDIM) + n * FDIM + k) * HID + col] : 0.0f;
  rep[idx] = (bf16)v;
}

// ========== K1: fused o@W1 -> relu -> @W2 -> +gumbel -> argmax -> masks =========
__global__ __launch_bounds__(32) void k1_assign(const float* __restrict__ x,
                                                const bf16* __restrict__ w1rep,
                                                const float* __restrict__ b1,
                                                const bf16* __restrict__ w2rep,
                                                const float* __restrict__ b2,
                                                unsigned char* __restrict__ assign,
                                                float* __restrict__ ig,
                                                float* __restrict__ og) {
  __shared__ __align__(16) bf16 hsh[16 * HID];  // h tile, bf16
  __shared__ float lsh[16 * 8];                 // logits+gumbel
  int lane = threadIdx.x;
  int half = lane >> 4, l16 = lane & 15;
  int rowbase = blockIdx.x * 16;
  int r = rowbase + l16;
  int b = r >> 7, n = r & 127;
  const float* obase = x + (size_t)b * XROW + 36 + n * FDIM;
  v16bf a = load_o_frag(obase, half);

  // GEMM1: 16x512 = A(16x32,K pad) @ W1(32x512), fused bias+relu -> LDS bf16
#pragma unroll 4
  for (int j = 0; j < 32; j++) {
    v16bf bb = *(const v16bf*)(w1rep + (j * 32 + lane) * 16);
    v8f c = {};
    c = wmma_bf16(a, bb, c);
    int col = j * 16 + l16;
    float bias = b1[col];
#pragma unroll
    for (int i = 0; i < 8; i++) {
      float h = fmaxf(c[i] + bias, 0.0f);
      hsh[(half ? 8 + i : i) * HID + col] = (bf16)h;
    }
  }
  __syncthreads();

  // GEMM2: logits(16x16, cols 0..7 valid) = h(16x512) @ W2pad(512x16)
  v8f c2 = {};
#pragma unroll 4
  for (int kt = 0; kt < 16; kt++) {
    const bf16* hp = hsh + l16 * HID + kt * 32 + (half ? 8 : 0);
    v8bf lo = *(const v8bf*)hp;
    v8bf hi = *(const v8bf*)(hp + 16);
    v16bf a2;
#pragma unroll
    for (int i = 0; i < 8; i++) { a2[i] = lo[i]; a2[8 + i] = hi[i]; }
    v16bf b2f = *(const v16bf*)(w2rep + (kt * 32 + lane) * 16);
    c2 = wmma_bf16(a2, b2f, c2);
  }
  int col = l16;
  if (col < 8) {
    float bias2 = b2[col];
#pragma unroll
    for (int i = 0; i < 8; i++) {
      int m = half ? 8 + i : i;
      uint32_t idx = (uint32_t)(rowbase + m) * 8u + (uint32_t)col;
      lsh[m * 8 + col] = c2[i] + bias2 + gumbel_at(idx);
    }
  }
  __syncthreads();

  if (lane < 16) {
    int m = lane;
    float best = lsh[m * 8];
    int av = 0;
#pragma unroll
    for (int g = 1; g < 8; g++) {
      float v = lsh[m * 8 + g];
      if (v > best) { best = v; av = g; }
    }
    int rr = rowbase + m, bb_ = rr >> 7, nn = rr & 127;
    assign[rr] = (unsigned char)av;
#pragma unroll
    for (int g = 0; g < 4; g++) ig[bb_ * 512 + g * 128 + nn] = (av == g) ? 1.0f : 0.0f;
#pragma unroll
    for (int g = 0; g < 4; g++) og[bb_ * 512 + g * 128 + nn] = (av == 4 + g) ? 1.0f : 0.0f;
  }
}

// ========== K1.5: deterministic ogroup_em[b][g*28+f] reduction ================
__global__ __launch_bounds__(128) void k15_oge(const float* __restrict__ x,
                                               const unsigned char* __restrict__ assign,
                                               float* __restrict__ oge) {
  __shared__ unsigned char ash[128];
  int b = blockIdx.x, t = threadIdx.x;
  ash[t] = assign[b * 128 + t];
  __syncthreads();
  if (t < 112) {
    int g = t / FDIM, f = t % FDIM;
    const float* xb = x + (size_t)b * XROW + 36;
    unsigned char want = (unsigned char)(4 + g);
    float s = 0.0f;
    for (int n = 0; n < 128; n++)
      if (ash[n] == want) s += xb[n * FDIM + f];
    oge[b * 112 + t] = s;
  }
}

// ========== K2: out = [self|ogroup_em] @ Wa_rows + ba  (M=4096,K=160,N=512) ====
__device__ inline float k2_a(const float* x, const float* oge, int b, int k) {
  if (k < 36) return x[(size_t)b * XROW + k];
  if (k < 148) return oge[b * 112 + (k - 36)];
  return 0.0f;
}

__global__ __launch_bounds__(256) void k2_base(const float* __restrict__ x,
                                               const float* __restrict__ oge,
                                               const bf16* __restrict__ k2rep,
                                               const float* __restrict__ ba,
                                               float* __restrict__ out) {
  int lane = threadIdx.x & 31, w = threadIdx.x >> 5;
  int half = lane >> 4, l16 = lane & 15;
  int Mbase = blockIdx.x * 128 + (w & 3) * 32;
  int Nbase = blockIdx.y * 64 + (w >> 2) * 32;
  v8f acc[2][2];
#pragma unroll
  for (int mi = 0; mi < 2; mi++)
#pragma unroll
    for (int ni = 0; ni < 2; ni++) {
      float bv = ba[Nbase + ni * 16 + l16];
#pragma unroll
      for (int i = 0; i < 8; i++) acc[mi][ni][i] = bv;
    }
#pragma unroll
  for (int kt = 0; kt < 5; kt++) {
    v16bf a[2];
#pragma unroll
    for (int mi = 0; mi < 2; mi++) {
      int b = Mbase + mi * 16 + l16;
#pragma unroll
      for (int i = 0; i < 16; i++) {
        int k = kt * 32 + ((i < 8) ? i : 8 + i) + (half ? 8 : 0);
        a[mi][i] = (bf16)k2_a(x, oge, b, k);
      }
    }
    v16bf bb[2];
#pragma unroll
    for (int ni = 0; ni < 2; ni++) {
      int j = (Nbase >> 4) + ni;
      bb[ni] = *(const v16bf*)(k2rep + ((kt * 32 + j) * 32 + lane) * 16);
    }
#pragma unroll
    for (int mi = 0; mi < 2; mi++)
#pragma unroll
      for (int ni = 0; ni < 2; ni++) acc[mi][ni] = wmma_bf16(a[mi], bb[ni], acc[mi][ni]);
  }
#pragma unroll
  for (int mi = 0; mi < 2; mi++)
#pragma unroll
    for (int ni = 0; ni < 2; ni++) {
      int row0 = Mbase + mi * 16 + half * 8;
      int colo = Nbase + ni * 16 + l16;
#pragma unroll
      for (int i = 0; i < 8; i++) out[(size_t)(row0 + i) * HID + colo] = acc[mi][ni][i];
    }
}

// ========== K3: out += sum_g (mask_g . o) @ Wa_i[g]   (the 60-GFLOP GEMM) ======
// Block: 256 threads = 8 waves; block tile 256x64; wave tile 64x32 (4x2 WMMA accs).
// Per node-step: async-stage o rows (28KB f32) + 4 groups' B fragments (16KB bf16)
// into double-buffered LDS; 32 WMMAs per wave per step.
#define K3_MB 256
__global__ __launch_bounds__(256) void k3_igroup(const float* __restrict__ x,
                                                 const unsigned char* __restrict__ assign,
                                                 const bf16* __restrict__ warep,
                                                 float* __restrict__ out) {
  __shared__ __align__(16) float abuf[2][K3_MB * FDIM];      // 2 x 28KB raw o rows
  __shared__ __align__(32) bf16  bbuf[2][4 * 2048];          // 2 x 16KB (4 g x 4KB)
  __shared__ __align__(16) unsigned char ashs[K3_MB * NNODE]; // 32KB assignments
  int tid = threadIdx.x;
  int lane = tid & 31, w = tid >> 5;
  int half = lane >> 4, l16 = lane & 15;
  int mw = w & 3, nw = w >> 2;
  int Mblock = blockIdx.x * K3_MB;
  int j0 = blockIdx.y * 4;  // first 16-col tile of this block (64 cols)

  // stage assignments for this block's rows (contiguous 32KB)
  {
    const uint4* src = (const uint4*)(assign + (size_t)Mblock * NNODE);
    uint4* dst = (uint4*)ashs;
#pragma unroll
    for (int i = 0; i < (K3_MB * NNODE / 16) / 256; i++) dst[tid + i * 256] = src[tid + i * 256];
  }

  // init accumulators from K2's output
  v8f acc[4][2];
#pragma unroll
  for (int mi = 0; mi < 4; mi++)
#pragma unroll
    for (int ni = 0; ni < 2; ni++) {
      int row0 = Mblock + mw * 64 + mi * 16 + half * 8;
      int colo = blockIdx.y * 64 + nw * 32 + ni * 16 + l16;
#pragma unroll
      for (int i = 0; i < 8; i++) acc[mi][ni][i] = out[(size_t)(row0 + i) * HID + colo];
    }

  // stage node n into buffer buf (each thread: 7 A-chunks for its row + 4 B-chunks)
  auto stage = [&](int buf, int n) {
    const float* ga = x + (size_t)(Mblock + tid) * XROW + 36 + n * FDIM;
    float* la = &abuf[buf][tid * FDIM];
#pragma unroll
    for (int q = 0; q < 7; q++) ASYNC_CP16(la + q * 4, ga + q * 4);
    int g = tid >> 6, c0 = tid & 63;
    const bf16* gb = warep + ((size_t)(g * 128 + n) * 32 + j0) * 32 * 16;
    bf16* lb = &bbuf[buf][g * 2048];
#pragma unroll
    for (int i = 0; i < 4; i++)
      ASYNC_CP16(lb + (c0 + i * 64) * 8, gb + (c0 + i * 64) * 8);
  };

  stage(0, 0);
  for (int n = 0; n < NNODE; n++) {
    int cur = n & 1;
    ASYNC_WAIT();          // stage(n) landed (this thread's ops)
    __syncthreads();       // all threads' stage(n) landed; buf[cur^1] fully consumed
    if (n + 1 < NNODE) stage(cur ^ 1, n + 1);

    // build raw (unmasked) A fragments + assignments for this node
    v8u raw[4];
    int av[4];
    const float* ab = &abuf[cur][0];
#pragma unroll
    for (int mi = 0; mi < 4; mi++) {
      int lrow = mw * 64 + mi * 16 + l16;
      raw[mi] = __builtin_bit_cast(v8u, load_o_frag(ab + lrow * FDIM, half));
      av[mi] = (int)ashs[lrow * NNODE + n];
    }
#pragma unroll
    for (int g = 0; g < 4; g++) {
      v16bf bb[2];
#pragma unroll
      for (int ni = 0; ni < 2; ni++) {
        int jl = nw * 2 + ni;
        bb[ni] = *(const v16bf*)(&bbuf[cur][g * 2048 + jl * 512 + lane * 16]);
      }
#pragma unroll
      for (int mi = 0; mi < 4; mi++) {
        uint32_t sel = (av[mi] == g) ? 0xFFFFFFFFu : 0u;
        v16bf am = __builtin_bit_cast(v16bf, raw[mi] & sel);
#pragma unroll
        for (int ni = 0; ni < 2; ni++) acc[mi][ni] = wmma_bf16(am, bb[ni], acc[mi][ni]);
      }
    }
  }

#pragma unroll
  for (int mi = 0; mi < 4; mi++)
#pragma unroll
    for (int ni = 0; ni < 2; ni++) {
      int row0 = Mblock + mw * 64 + mi * 16 + half * 8;
      int colo = blockIdx.y * 64 + nw * 32 + ni * 16 + l16;
#pragma unroll
      for (int i = 0; i < 8; i++) out[(size_t)(row0 + i) * HID + colo] = acc[mi][ni][i];
    }
}

// ============================== launch =========================================
extern "C" void kernel_launch(void* const* d_in, const int* in_sizes, int n_in,
                              void* d_out, int out_size, void* d_ws, size_t ws_size,
                              hipStream_t stream) {
  const float* x  = (const float*)d_in[0];
  const float* W1 = (const float*)d_in[1];
  const float* b1 = (const float*)d_in[2];
  const float* W2 = (const float*)d_in[3];
  const float* b2 = (const float*)d_in[4];
  const float* Wa = (const float*)d_in[5];
  const float* ba = (const float*)d_in[6];

  float* out = (float*)d_out;
  float* ig = out + (size_t)BATCH * HID;
  float* og = ig + (size_t)BATCH * 4 * NNODE;

  char* ws = (char*)d_ws;
  unsigned char* assign = (unsigned char*)(ws + WS_ASSIGN);
  float* oge   = (float*)(ws + WS_OGE);
  bf16* w1rep  = (bf16*)(ws + WS_W1REP);
  bf16* w2rep  = (bf16*)(ws + WS_W2REP);
  bf16* k2rep  = (bf16*)(ws + WS_K2REP);
  bf16* warep  = (bf16*)(ws + WS_WAREP);

  repack_w1<<<(32 * 32 * 16 + 255) / 256, 256, 0, stream>>>(W1, w1rep);
  repack_w2<<<(16 * 32 * 16 + 255) / 256, 256, 0, stream>>>(W2, w2rep);
  repack_k2<<<(5 * 32 * 32 * 16 + 255) / 256, 256, 0, stream>>>(Wa, k2rep);
  repack_wa<<<(4 * 128 * 32 * 32 * 16) / 256, 256, 0, stream>>>(Wa, warep);

  k1_assign<<<NROWS / 16, 32, 0, stream>>>(x, w1rep, b1, w2rep, b2, assign, ig, og);
  k15_oge<<<BATCH, 128, 0, stream>>>(x, assign, oge);

  dim3 g2(BATCH / 128, HID / 64);
  k2_base<<<g2, 256, 0, stream>>>(x, oge, k2rep, ba, out);

  dim3 g3(BATCH / K3_MB, HID / 64);
  k3_igroup<<<g3, 256, 0, stream>>>(x, assign, warep, out);
}
